// Convolution_88235808129501
// MI455X (gfx1250) — matrix operand order repaired and checked by reading, not compiled, not run on hardware
//
#include <hip/hip_runtime.h>

// ---- problem shape ---------------------------------------------------------
#define CIN   128
#define HHH   64
#define WWW   64
#define COUT  256
#define OHH   62
#define OWW   62
#define NBATCH 32
#define PLANE (OHH * OWW)           // 3844
#define KDIM  (CIN * 9)             // 1152 = 36 * 32
#define NDIM  (NBATCH * PLANE)      // 123008 = 961 * 128

// ---- GEMM tiling -----------------------------------------------------------
#define BM 128
#define BN 128
#define BK 32
#define LDK 40                      // padded LDS row: 80B (16B-aligned, conflict-free)
#define KSTEPS (KDIM / BK)          // 36

typedef _Float16 half8 __attribute__((ext_vector_type(8)));
typedef _Float16 v16h  __attribute__((ext_vector_type(16)));
typedef float    v8f   __attribute__((ext_vector_type(8)));
typedef int      v4i   __attribute__((ext_vector_type(4)));

#define ASG __attribute__((address_space(1)))
#define ASL __attribute__((address_space(3)))

#if __has_builtin(__builtin_amdgcn_global_load_async_to_lds_b128)
#define USE_ASYNC_LDS 1
#endif

__device__ __forceinline__ void wait_async_lds() {
#ifdef USE_ASYNC_LDS
#if __has_builtin(__builtin_amdgcn_s_wait_asynccnt)
    __builtin_amdgcn_s_wait_asynccnt(0);
#else
    asm volatile("s_wait_asynccnt 0x0" ::: "memory");
#endif
#endif
}

// ---- pre-pass: weights f32 -> f16 into workspace (1152 blocks x 256) -------
__global__ __launch_bounds__(256) void wcvt_f32_to_f16(
    const float* __restrict__ w, _Float16* __restrict__ o) {
    const int i = blockIdx.x * 256 + threadIdx.x;   // exact cover of 294912
    o[i] = (_Float16)w[i];
}

// ---- main implicit-GEMM conv kernel ----------------------------------------
__global__ __launch_bounds__(256) void conv_implicit_gemm_wmma(
    const float*    __restrict__ in,    // [32,128,64,64] f32
    const _Float16* __restrict__ wh,    // [256,1152] f16 (pre-converted)
    const float*    __restrict__ bias,  // [256]
    float*          __restrict__ out)   // [32,256,62,62]
{
    __shared__ _Float16 As[2][BM * LDK];   // weights tile  [m][k]
    __shared__ _Float16 Bs[2][BN * LDK];   // im2col tile   [n][k]

    const int tid  = threadIdx.x;
    const int wave = tid >> 5;
    const int lane = tid & 31;
    const int h    = lane >> 4;
    const int l16  = lane & 15;
    const int wm   = wave >> 1;            // 0..3 : 32-row M strip
    const int wn   = wave & 1;             // 0..1 : 64-col N strip

    const int n0 = blockIdx.x * BN;        // 961 blocks (exact)
    const int m0 = blockIdx.y * BM;        // 2 blocks (exact)

    // ---- loop-invariant B-staging state: n = tid&127 is fixed per thread --
    const int bn   = tid & 127;
    const int kb   = (tid >> 7) << 4;      // 0 or 16 (k sub-block)
    const int ng   = n0 + bn;
    const int bb   = ng / PLANE;
    const int rm   = ng - bb * PLANE;
    const int oi   = rm / OWW;
    const int oj   = rm - oi * OWW;
    const float* bsrc = in + (size_t)bb * (CIN * HHH * WWW) + oi * WWW + oj;

    // ---- A staging: thread copies 2 x b128 (8 f16) chunks per step --------
    auto stageA = [&](int s, int buf) {
        #pragma unroll
        for (int it = 0; it < 2; ++it) {
            const int chunk = tid + 256 * it;          // 0..511
            const int row   = chunk >> 2;              // 0..127
            const int q     = chunk & 3;               // b128 within row
            const _Float16* src = wh + (size_t)(m0 + row) * KDIM + s * BK + q * 8;
            _Float16*       dst = &As[buf][row * LDK + q * 8];
#ifdef USE_ASYNC_LDS
            __builtin_amdgcn_global_load_async_to_lds_b128(
                (ASG v4i*)src, (ASL v4i*)dst, 0, 0);
#else
            v4i t = *(const v4i*)src;
            *(v4i*)dst = t;
#endif
        }
    };

    // ---- B staging: global f32 gather -> regs (prefetch), regs -> LDS f16 -
    auto loadB = [&](int s, float* br) {
        #pragma unroll
        for (int it = 0; it < 16; ++it) {
            const int kg = s * BK + kb + it;
            const int c  = (int)((unsigned)kg / 9u);
            const int t9 = kg - c * 9;
            const int u  = (int)((unsigned)t9 / 3u);
            const int v  = t9 - u * 3;
            br[it] = bsrc[(size_t)c * (HHH * WWW) + u * WWW + v];
        }
    };
    auto storeB = [&](int buf, const float* br) {
        half8 p0, p1;
        #pragma unroll
        for (int e = 0; e < 8; ++e) {
            p0[e] = (_Float16)br[e];
            p1[e] = (_Float16)br[8 + e];
        }
        *(half8*)&Bs[buf][bn * LDK + kb]     = p0;   // 16B aligned
        *(half8*)&Bs[buf][bn * LDK + kb + 8] = p1;
    };

    v8f acc[2][4] = {};

    // ---- compute: preload all fragments, then 8 back-to-back WMMAs --------
    auto compute = [&](int buf) {
        v16h a[2];
        v16h bf[4];
        #pragma unroll
        for (int mf = 0; mf < 2; ++mf) {
            const _Float16* ar = &As[buf][(wm * 32 + mf * 16 + l16) * LDK + h * 16];
            half8 lo = *(const half8*)ar;
            half8 hi = *(const half8*)(ar + 8);
            #pragma unroll
            for (int e = 0; e < 8; ++e) { a[mf][e] = lo[e]; a[mf][8 + e] = hi[e]; }
        }
        #pragma unroll
        for (int nt = 0; nt < 4; ++nt) {
            bf[nt] = *(const v16h*)&Bs[buf][(wn * 64 + nt * 16 + l16) * LDK + h * 16];
        }
        // single dscnt wait covers all 12 ds_load_b128, then straight-line WMMA
        #pragma unroll
        for (int nt = 0; nt < 4; ++nt) {
            acc[0][nt] = __builtin_amdgcn_wmma_f32_16x16x32_f16(
                false, a[0], false, bf[nt], (short)0, acc[0][nt], false, false);
            acc[1][nt] = __builtin_amdgcn_wmma_f32_16x16x32_f16(
                false, a[1], false, bf[nt], (short)0, acc[1][nt], false, false);
        }
    };

    // ---- software pipeline: prefetch s+1 while computing s ----------------
    float br[16];
    loadB(0, br);
    stageA(0, 0);
    storeB(0, br);
    wait_async_lds();
    __syncthreads();

    for (int s = 0; s < KSTEPS; ++s) {
        const int cur = s & 1;
        if (s + 1 < KSTEPS) {
            loadB(s + 1, br);            // global loads in flight over compute
            stageA(s + 1, 1 - cur);      // async DMA into the other buffer
        }
        compute(cur);
        if (s + 1 < KSTEPS) storeB(1 - cur, br);
        wait_async_lds();
        __syncthreads();
    }

    // ---- epilogue: D layout VGPR r -> row (r + 8h), col l16; add bias -----
    #pragma unroll
    for (int nt = 0; nt < 4; ++nt) {
        const int ngo = n0 + wn * 64 + nt * 16 + l16;
        const int b2  = ngo / PLANE;
        const int r2  = ngo - b2 * PLANE;
        const int i2  = r2 / OWW;
        const int j2  = r2 - i2 * OWW;
        float* obase = out + (size_t)b2 * (COUT * PLANE) + i2 * OWW + j2;
        #pragma unroll
        for (int mf = 0; mf < 2; ++mf) {
            #pragma unroll
            for (int r = 0; r < 8; ++r) {
                const int o = m0 + wm * 32 + mf * 16 + h * 8 + r;
                obase[(size_t)o * PLANE] = acc[mf][nt][r] + bias[o];
            }
        }
    }
}

extern "C" void kernel_launch(void* const* d_in, const int* in_sizes, int n_in,
                              void* d_out, int out_size, void* d_ws, size_t ws_size,
                              hipStream_t stream) {
    const float* in   = (const float*)d_in[0];   // [32,128,64,64]
    const float* wt   = (const float*)d_in[1];   // [256,128,3,3]
    const float* bias = (const float*)d_in[2];   // [256,1]
    float* out        = (float*)d_out;

    _Float16* wh = (_Float16*)d_ws;              // needs 294912*2 = 576 KB scratch

    // pass 1: weights f32 -> f16 (1.2 MB read, runs in ~microseconds)
    wcvt_f32_to_f16<<<(COUT * KDIM) / 256, 256, 0, stream>>>(wt, wh);

    // pass 2: implicit GEMM, M=256 N=123008 K=1152
    dim3 grid(NDIM / BN, COUT / BM);             // (961, 2)
    conv_implicit_gemm_wmma<<<grid, 256, 0, stream>>>(in, wh, bias, out);
}
